// TransformerEncoderLayer_3693671875072
// MI455X (gfx1250) — matrix operand order repaired and checked by reading, not compiled, not run on hardware
//
#include <hip/hip_runtime.h>
#include <cstdint>
#include <cstddef>

typedef _Float16 f16;
typedef _Float16 v16h __attribute__((ext_vector_type(16)));
typedef _Float16 v8h  __attribute__((ext_vector_type(8)));
typedef float    v8f  __attribute__((ext_vector_type(8)));
typedef unsigned int v4u __attribute__((ext_vector_type(4)));
typedef int          v8i __attribute__((ext_vector_type(8)));
typedef int          v4i __attribute__((ext_vector_type(4)));

#define D_MODEL 768
#define HEADS   12
#define DHEAD   64
#define D_FF    3072
#define BATCH   2
#define SEQ     2048
#define ROWS    (BATCH*SEQ)   // 4096

#ifndef __has_builtin
#define __has_builtin(x) 0
#endif
#if __has_builtin(__builtin_amdgcn_tensor_load_to_lds) && __has_builtin(__builtin_amdgcn_s_wait_tensorcnt)
#define HAVE_TDM 1
#endif

static __device__ __forceinline__ v16h cat8(v8h lo, v8h hi){
  return __builtin_shufflevector(lo, hi, 0,1,2,3,4,5,6,7,8,9,10,11,12,13,14,15);
}
static __device__ __forceinline__ v8f wmma32(v16h a, v16h b, v8f c){
  // D = A(16x32 f16) * B(32x16 f16) + C(16x16 f32)
  return __builtin_amdgcn_wmma_f32_16x16x32_f16(false, a, false, b, (short)0, c, false, false);
}
static __device__ __forceinline__ float gelu_tanh_f(float x){
  const float c = 0.7978845608028654f; // sqrt(2/pi)
  float u = c * (x + 0.044715f * x * x * x);
  return 0.5f * x * (1.0f + tanhf(u));
}

#ifdef HAVE_TDM
// Issue a TDM DMA: copy a [64 rows x 32 K-elems] f16 tile of Bt (row stride K elems)
// starting at global byte address `gaddr` into LDS at byte offset `lds_addr`.
// D# layout per CDNA5 ISA ch.8 (group0 128b, group1 256b; groups 2/3 unused for 2D).
static __device__ __forceinline__ void tdm_load_b_tile(unsigned int lds_addr,
                                                       unsigned long long gaddr,
                                                       int K_, int N_){
  v4u g0;
  g0[0] = 1u;                                           // count=1, user descriptor
  g0[1] = lds_addr;                                     // lds_addr [63:32]
  g0[2] = (unsigned int)gaddr;                          // global_addr [95:64]
  g0[3] = (unsigned int)((gaddr >> 32) & 0x01FFFFFFu)   // global_addr [120:96]
        | (2u << 30);                                   // type=2 ("image") [127:126]
  v8i g1;
  g1[0] = (int)(1u << 16);                              // wg_mask=0, data_size=1 (2 bytes)
  g1[1] = (int)((unsigned)(K_ & 0xFFFF) << 16);         // tensor_dim0 low16 at [63:48]
  g1[2] = (int)(((unsigned)K_ >> 16) & 0xFFFFu)         // tensor_dim0 high16
        | (int)((unsigned)(N_ & 0xFFFF) << 16);         // tensor_dim1 low16
  g1[3] = (int)(((unsigned)N_ >> 16) & 0xFFFFu)         // tensor_dim1 high16
        | (int)(32u << 16);                             // tile_dim0 = 32
  g1[4] = 64;                                           // tile_dim1 = 64, tile_dim2 = 0
  g1[5] = K_;                                           // tensor_dim0_stride low32 (elems)
  g1[6] = 0;
  g1[7] = 0;
  v4i z4 = {0,0,0,0};
#if defined(__clang_major__) && __clang_major__ >= 23
  v8i z8 = {0,0,0,0,0,0,0,0};
  __builtin_amdgcn_tensor_load_to_lds(g0, g1, z4, z4, z8, 0);
#else
  __builtin_amdgcn_tensor_load_to_lds(g0, g1, z4, z4, 0);
#endif
}
#endif

// ---------------- weight convert: fp32 [K,N] -> f16 [N,K] (transposed) ----------------
__global__ __launch_bounds__(256) void wcvt_t_kernel(const float* __restrict__ W,
                                                     f16* __restrict__ Wt,
                                                     int K, int N){
  int idx = blockIdx.x * 256 + threadIdx.x;
  if (idx >= K * N) return;
  int k = idx / N, n = idx - k * N;
  Wt[(size_t)n * K + k] = (f16)W[idx];
}

// ---------------- LayerNorm over rows of 768 -> f16 (optional passthrough) ----------------
__global__ __launch_bounds__(256) void ln_rows_kernel(const float* __restrict__ x,
                                                      const float* __restrict__ g,
                                                      const float* __restrict__ b,
                                                      f16* __restrict__ out,
                                                      const int* __restrict__ iterp){
  int row = blockIdx.x;
  const float* xr = x + (size_t)row * D_MODEL;
  int t = threadIdx.x;
  float v0 = xr[t], v1 = xr[t+256], v2 = xr[t+512];
  float s  = v0 + v1 + v2;
  float s2 = v0*v0 + v1*v1 + v2*v2;
  #pragma unroll
  for (int m = 16; m; m >>= 1){ s += __shfl_xor(s, m, 32); s2 += __shfl_xor(s2, m, 32); }
  __shared__ float sm[8], sq[8];
  if ((t & 31) == 0){ sm[t>>5] = s; sq[t>>5] = s2; }
  __syncthreads();
  float ts = 0.f, tq = 0.f;
  #pragma unroll
  for (int i = 0; i < 8; i++){ ts += sm[i]; tq += sq[i]; }
  float mu   = ts * (1.0f/D_MODEL);
  float var  = tq * (1.0f/D_MODEL) - mu*mu;
  float rstd = rsqrtf(var + 1e-6f);
  bool plain = (iterp != nullptr) && (iterp[0] == 0);
  f16* orow = out + (size_t)row * D_MODEL;
  if (plain){
    orow[t] = (f16)v0; orow[t+256] = (f16)v1; orow[t+512] = (f16)v2;
  } else {
    orow[t]     = (f16)((v0-mu)*rstd*g[t]     + b[t]);
    orow[t+256] = (f16)((v1-mu)*rstd*g[t+256] + b[t+256]);
    orow[t+512] = (f16)((v2-mu)*rstd*g[t+512] + b[t+512]);
  }
}

// ---------------- WMMA GEMM: C = epilogue(A[M,K] x Bt[N,K]^T + bias) ----------------
#define EP_H16       0   // store f16 row-major
#define EP_H16_SCALE 1   // (acc+bias)*scale -> f16  (query)
#define EP_VT        2   // f16 into [B][H][DHEAD][SEQ] transposed-per-head (value)
#define EP_GELU      3   // tanh-GELU -> f16
#define EP_F32_ADD   4   // acc+bias+addsrc -> f32

__global__ __launch_bounds__(256) void gemm_wmma_kernel(
    const f16* __restrict__ A, int lda,
    const f16* __restrict__ Bt,      // [N][K] transposed weights
    const float* __restrict__ bias,  // [N]
    int M, int N, int K,
    int ep, float scale,
    const float* __restrict__ addsrc,
    float* __restrict__ outF,
    f16* __restrict__ outH){
  int lane = threadIdx.x & 31;
  int w    = threadIdx.x >> 5;
  int ml = lane & 15, hi = lane >> 4;
  int m0 = blockIdx.y * 128 + w * 16;   // 8 waves x 16 rows
  int n0 = blockIdx.x * 64;             // 4 accumulators x 16 cols

  v8f acc0 = {}, acc1 = {}, acc2 = {}, acc3 = {};
  // A fragment lane layout: row = m0+ml; K {hi*16..+15} contiguous
  const f16* arow = A + (size_t)(m0 + ml) * lda + hi * 16;

#ifdef HAVE_TDM
  // Block-shared B tile [64 n-rows][32 k] staged by the Tensor Data Mover, double buffered.
  __shared__ f16 Btile[2][64][32];
  const unsigned long long gbase =
      (unsigned long long)(uintptr_t)(Bt + (size_t)n0 * K);
  const unsigned int ldsb0 = (unsigned int)(uintptr_t)(&Btile[0][0][0]);
  const unsigned int ldsb1 = (unsigned int)(uintptr_t)(&Btile[1][0][0]);
  if (w == 0){
    tdm_load_b_tile(ldsb0, gbase, K, N);       // tile for k0 = 0
    __builtin_amdgcn_s_wait_tensorcnt(0);
  }
  __syncthreads();
  int buf = 0;
  for (int k0 = 0; k0 < K; k0 += 32){
    if (w == 0 && (k0 + 32) < K){
      tdm_load_b_tile(buf ? ldsb0 : ldsb1, gbase + (size_t)(k0 + 32) * 2, K, N);
    }
    v16h a = cat8(*(const v8h*)(arow + k0), *(const v8h*)(arow + k0 + 8));
    {
      const f16* bp0 = &Btile[buf][ 0 + ml][hi * 16];
      const f16* bp1 = &Btile[buf][16 + ml][hi * 16];
      const f16* bp2 = &Btile[buf][32 + ml][hi * 16];
      const f16* bp3 = &Btile[buf][48 + ml][hi * 16];
      acc0 = wmma32(a, cat8(*(const v8h*)bp0, *(const v8h*)(bp0 + 8)), acc0);
      acc1 = wmma32(a, cat8(*(const v8h*)bp1, *(const v8h*)(bp1 + 8)), acc1);
      acc2 = wmma32(a, cat8(*(const v8h*)bp2, *(const v8h*)(bp2 + 8)), acc2);
      acc3 = wmma32(a, cat8(*(const v8h*)bp3, *(const v8h*)(bp3 + 8)), acc3);
    }
    __builtin_prefetch(arow + k0 + 64, 0, 3);
    __syncthreads();                            // all waves done with `buf`
    if (w == 0) __builtin_amdgcn_s_wait_tensorcnt(0);
    __syncthreads();                            // next tile visible to all waves
    buf ^= 1;
  }
#else
  // Fallback: per-lane direct global B fragment loads (16 contiguous K per lane)
  const f16* b0p = Bt + (size_t)(n0 +  0 + ml) * K + hi * 16;
  const f16* b1p = Bt + (size_t)(n0 + 16 + ml) * K + hi * 16;
  const f16* b2p = Bt + (size_t)(n0 + 32 + ml) * K + hi * 16;
  const f16* b3p = Bt + (size_t)(n0 + 48 + ml) * K + hi * 16;
  for (int k0 = 0; k0 < K; k0 += 32){
    v16h a  = cat8(*(const v8h*)(arow + k0), *(const v8h*)(arow + k0 + 8));
    v16h b0 = cat8(*(const v8h*)(b0p + k0), *(const v8h*)(b0p + k0 + 8));
    v16h b1 = cat8(*(const v8h*)(b1p + k0), *(const v8h*)(b1p + k0 + 8));
    v16h b2 = cat8(*(const v8h*)(b2p + k0), *(const v8h*)(b2p + k0 + 8));
    v16h b3 = cat8(*(const v8h*)(b3p + k0), *(const v8h*)(b3p + k0 + 8));
    acc0 = wmma32(a, b0, acc0);
    acc1 = wmma32(a, b1, acc1);
    acc2 = wmma32(a, b2, acc2);
    acc3 = wmma32(a, b3, acc3);
    __builtin_prefetch(arow + k0 + 64, 0, 3);
  }
#endif

  v8f accs[4] = {acc0, acc1, acc2, acc3};
  #pragma unroll
  for (int j = 0; j < 4; j++){
    int n = n0 + j*16 + ml;
    float bv = bias ? bias[n] : 0.0f;
    #pragma unroll
    for (int r = 0; r < 8; r++){
      int m = m0 + r + hi*8;           // C layout: lane holds col n, rows r+8*hi
      float v = accs[j][r] + bv;
      if (ep == EP_H16){
        outH[(size_t)m * N + n] = (f16)v;
      } else if (ep == EP_H16_SCALE){
        outH[(size_t)m * N + n] = (f16)(v * scale);
      } else if (ep == EP_VT){
        int bb = m >> 11, ss = m & (SEQ - 1);
        int hh = n >> 6,  dd = n & (DHEAD - 1);
        outH[((size_t)(bb*HEADS + hh) * DHEAD + dd) * SEQ + ss] = (f16)v;
      } else if (ep == EP_GELU){
        outH[(size_t)m * N + n] = (f16)gelu_tanh_f(v);
      } else { // EP_F32_ADD
        outF[(size_t)m * N + n] = v + addsrc[(size_t)m * N + n];
      }
    }
  }
}

// ---------------- fused flash attention ----------------
// grid (SEQ/64, HEADS, BATCH), block 128 (4 waves, each wave = 16 query rows)
__global__ __launch_bounds__(128) void attn_kernel(
    const f16* __restrict__ qh,   // [ROWS][D_MODEL], pre-scaled by 1/sqrt(dh)
    const f16* __restrict__ kh,   // [ROWS][D_MODEL]
    const f16* __restrict__ vt,   // [B][H][DHEAD][SEQ]
    const unsigned char* __restrict__ mask, // [B*SEQ] key mask (true -> -1e18)
    f16* __restrict__ ctx){       // [ROWS][D_MODEL]
  int lane = threadIdx.x & 31;
  int w    = threadIdx.x >> 5;
  int ml = lane & 15, hi = lane >> 4;
  int q0 = blockIdx.x * 64 + w * 16;
  int h  = blockIdx.y;
  int b  = blockIdx.z;
  __shared__ f16 P[4][16][32];   // per-wave probability staging (C-layout -> A-layout)

  const f16* qrow = qh + (size_t)(b*SEQ + q0 + ml) * D_MODEL + h*DHEAD + hi*16;
  v16h qa0 = cat8(*(const v8h*)(qrow),      *(const v8h*)(qrow + 8)); // dh 0..31
  v16h qa1 = cat8(*(const v8h*)(qrow + 32), *(const v8h*)(qrow + 40)); // dh 32..63

  float mrun[8], lrun[8];
  #pragma unroll
  for (int r = 0; r < 8; r++){ mrun[r] = -3.0e38f; lrun[r] = 0.0f; }
  v8f O0 = {}, O1 = {}, O2 = {}, O3 = {};

  const unsigned char* mrow = mask + (size_t)b * SEQ;
  const f16* vbase = vt + (size_t)(b*HEADS + h) * DHEAD * SEQ;

  for (int kt = 0; kt < SEQ; kt += 32){
    v8f sc0 = {}, sc1 = {};
    {
      const f16* krow0 = kh + (size_t)(b*SEQ + kt + ml) * D_MODEL + h*DHEAD + hi*16;
      v16h kb = cat8(*(const v8h*)(krow0),      *(const v8h*)(krow0 + 8));
      sc0 = wmma32(qa0, kb, sc0);
      kb = cat8(*(const v8h*)(krow0 + 32), *(const v8h*)(krow0 + 40));
      sc0 = wmma32(qa1, kb, sc0);
      const f16* krow1 = krow0 + 16 * D_MODEL;
      kb = cat8(*(const v8h*)(krow1),      *(const v8h*)(krow1 + 8));
      sc1 = wmma32(qa0, kb, sc1);
      kb = cat8(*(const v8h*)(krow1 + 32), *(const v8h*)(krow1 + 40));
      sc1 = wmma32(qa1, kb, sc1);
    }
    // key mask: lane owns one key column per accumulator
    if (mrow[kt + ml]){
      #pragma unroll
      for (int r = 0; r < 8; r++) sc0[r] = -1.0e18f;
    }
    if (mrow[kt + 16 + ml]){
      #pragma unroll
      for (int r = 0; r < 8; r++) sc1[r] = -1.0e18f;
    }
    // row max over 32 key cols (16-lane xor reduction within each half)
    float rmax[8];
    #pragma unroll
    for (int r = 0; r < 8; r++) rmax[r] = fmaxf(sc0[r], sc1[r]);
    #pragma unroll
    for (int ms = 1; ms < 16; ms <<= 1){
      #pragma unroll
      for (int r = 0; r < 8; r++) rmax[r] = fmaxf(rmax[r], __shfl_xor(rmax[r], ms, 32));
    }
    float mnew[8], sfac[8], rsum[8];
    #pragma unroll
    for (int r = 0; r < 8; r++){
      mnew[r] = fmaxf(mrun[r], rmax[r]);
      sfac[r] = __expf(mrun[r] - mnew[r]);
      sc0[r]  = __expf(sc0[r] - mnew[r]);
      sc1[r]  = __expf(sc1[r] - mnew[r]);
      rsum[r] = sc0[r] + sc1[r];
    }
    #pragma unroll
    for (int ms = 1; ms < 16; ms <<= 1){
      #pragma unroll
      for (int r = 0; r < 8; r++) rsum[r] += __shfl_xor(rsum[r], ms, 32);
    }
    #pragma unroll
    for (int r = 0; r < 8; r++){
      lrun[r] = lrun[r]*sfac[r] + rsum[r];
      mrun[r] = mnew[r];
      O0[r] *= sfac[r]; O1[r] *= sfac[r]; O2[r] *= sfac[r]; O3[r] *= sfac[r];
    }
    // stage P (16 q-rows x 32 keys) through LDS to re-layout C -> A fragment
    #pragma unroll
    for (int r = 0; r < 8; r++){
      int row = r + hi*8;
      P[w][row][ml]      = (f16)sc0[r];
      P[w][row][16 + ml] = (f16)sc1[r];
    }
    const f16* pr = &P[w][ml][hi*16];   // wave-private; DS ops in-order within wave
    v16h pa = cat8(*(const v8h*)pr, *(const v8h*)(pr + 8));
    // O += P x V  (V transposed per head: contiguous keys per lane)
    const f16* vr0 = vbase + (size_t)( 0 + ml) * SEQ + kt + hi*16;
    const f16* vr1 = vbase + (size_t)(16 + ml) * SEQ + kt + hi*16;
    const f16* vr2 = vbase + (size_t)(32 + ml) * SEQ + kt + hi*16;
    const f16* vr3 = vbase + (size_t)(48 + ml) * SEQ + kt + hi*16;
    O0 = wmma32(pa, cat8(*(const v8h*)vr0, *(const v8h*)(vr0 + 8)), O0);
    O1 = wmma32(pa, cat8(*(const v8h*)vr1, *(const v8h*)(vr1 + 8)), O1);
    O2 = wmma32(pa, cat8(*(const v8h*)vr2, *(const v8h*)(vr2 + 8)), O2);
    O3 = wmma32(pa, cat8(*(const v8h*)vr3, *(const v8h*)(vr3 + 8)), O3);
  }
  #pragma unroll
  for (int r = 0; r < 8; r++){
    float inv = 1.0f / lrun[r];
    int srow = q0 + r + hi*8;
    f16* crow = ctx + (size_t)(b*SEQ + srow) * D_MODEL + h*DHEAD;
    crow[ 0 + ml] = (f16)(O0[r] * inv);
    crow[16 + ml] = (f16)(O1[r] * inv);
    crow[32 + ml] = (f16)(O2[r] * inv);
    crow[48 + ml] = (f16)(O3[r] * inv);
  }
}

extern "C" void kernel_launch(void* const* d_in, const int* in_sizes, int n_in,
                              void* d_out, int out_size, void* d_ws, size_t ws_size,
                              hipStream_t stream){
  (void)in_sizes; (void)n_in; (void)out_size;
  const int*   iterp  = (const int*)  d_in[0];
  /* d_in[1] = query: unused by the reference computation */
  const float* inputs = (const float*)d_in[2];
  const unsigned char* maskp = (const unsigned char*)d_in[3];
  const float* Wq = (const float*)d_in[4];  const float* bq = (const float*)d_in[5];
  const float* Wk = (const float*)d_in[6];  const float* bk = (const float*)d_in[7];
  const float* Wv = (const float*)d_in[8];  const float* bv = (const float*)d_in[9];
  const float* Wo = (const float*)d_in[10]; const float* bo = (const float*)d_in[11];
  const float* ln1g = (const float*)d_in[12]; const float* ln1b = (const float*)d_in[13];
  const float* W1 = (const float*)d_in[14]; const float* b1 = (const float*)d_in[15];
  const float* W2 = (const float*)d_in[16]; const float* b2 = (const float*)d_in[17];
  const float* ln2g = (const float*)d_in[18]; const float* ln2b = (const float*)d_in[19];

  char* ws = (char*)d_ws;
  size_t off = 0;
  auto carve = [&](size_t bytes)->char*{
    char* p = ws + off;
    off += (bytes + 255) & ~(size_t)255;
    return p;
  };
  f16*   WqT    = (f16*)carve((size_t)D_MODEL*D_MODEL*2);
  f16*   WkT    = (f16*)carve((size_t)D_MODEL*D_MODEL*2);
  f16*   WvT    = (f16*)carve((size_t)D_MODEL*D_MODEL*2);
  f16*   WoT    = (f16*)carve((size_t)D_MODEL*D_MODEL*2);
  f16*   W1T    = (f16*)carve((size_t)D_MODEL*D_FF*2);
  f16*   W2T    = (f16*)carve((size_t)D_FF*D_MODEL*2);
  f16*   xh     = (f16*)carve((size_t)ROWS*D_MODEL*2);
  f16*   qh     = (f16*)carve((size_t)ROWS*D_MODEL*2);
  f16*   kh     = (f16*)carve((size_t)ROWS*D_MODEL*2);
  f16*   vth    = (f16*)carve((size_t)ROWS*D_MODEL*2);
  float* out32  = (float*)carve((size_t)ROWS*D_MODEL*4);
  f16*   interh = (f16*)carve((size_t)ROWS*D_FF*2);
  if (off > ws_size) return;   // insufficient workspace (deterministic no-op)
  f16* ctx = xh;   // xh dead after V projection
  f16* hh  = qh;   // qh dead after attention

  // 1) weights fp32 -> f16 transposed
  int nSq = D_MODEL*D_MODEL;
  wcvt_t_kernel<<<(nSq+255)/256, 256, 0, stream>>>(Wq, WqT, D_MODEL, D_MODEL);
  wcvt_t_kernel<<<(nSq+255)/256, 256, 0, stream>>>(Wk, WkT, D_MODEL, D_MODEL);
  wcvt_t_kernel<<<(nSq+255)/256, 256, 0, stream>>>(Wv, WvT, D_MODEL, D_MODEL);
  wcvt_t_kernel<<<(nSq+255)/256, 256, 0, stream>>>(Wo, WoT, D_MODEL, D_MODEL);
  int nFf = D_MODEL*D_FF;
  wcvt_t_kernel<<<(nFf+255)/256, 256, 0, stream>>>(W1, W1T, D_MODEL, D_FF);
  wcvt_t_kernel<<<(nFf+255)/256, 256, 0, stream>>>(W2, W2T, D_FF, D_MODEL);

  // 2) pre-LN (skipped if iter==0)
  ln_rows_kernel<<<ROWS, 256, 0, stream>>>(inputs, ln1g, ln1b, xh, iterp);

  // 3) Q/K/V projections
  dim3 g768(D_MODEL/64, ROWS/128, 1);
  gemm_wmma_kernel<<<g768, 256, 0, stream>>>(xh, D_MODEL, WqT, bq, ROWS, D_MODEL, D_MODEL,
                                             EP_H16_SCALE, 0.125f, nullptr, nullptr, qh);
  gemm_wmma_kernel<<<g768, 256, 0, stream>>>(xh, D_MODEL, WkT, bk, ROWS, D_MODEL, D_MODEL,
                                             EP_H16, 1.0f, nullptr, nullptr, kh);
  gemm_wmma_kernel<<<g768, 256, 0, stream>>>(xh, D_MODEL, WvT, bv, ROWS, D_MODEL, D_MODEL,
                                             EP_VT, 1.0f, nullptr, nullptr, vth);

  // 4) fused attention -> ctx
  attn_kernel<<<dim3(SEQ/64, HEADS, BATCH), 128, 0, stream>>>(qh, kh, vth, maskp, ctx);

  // 5) output projection + residual -> out32 (fp32)
  gemm_wmma_kernel<<<g768, 256, 0, stream>>>(ctx, D_MODEL, WoT, bo, ROWS, D_MODEL, D_MODEL,
                                             EP_F32_ADD, 1.0f, inputs, out32, nullptr);

  // 6) LN2 -> hh (f16)
  ln_rows_kernel<<<ROWS, 256, 0, stream>>>(out32, ln2g, ln2b, hh, nullptr);

  // 7) FFN up + GELU
  dim3 gff(D_FF/64, ROWS/128, 1);
  gemm_wmma_kernel<<<gff, 256, 0, stream>>>(hh, D_MODEL, W1T, b1, ROWS, D_FF, D_MODEL,
                                            EP_GELU, 1.0f, nullptr, nullptr, interh);

  // 8) FFN down + bias + residual -> d_out (fp32)
  gemm_wmma_kernel<<<g768, 256, 0, stream>>>(interh, D_FF, W2T, b2, ROWS, D_MODEL, D_FF,
                                             EP_F32_ADD, 1.0f, out32, (float*)d_out, nullptr);
}
